// MultiHeadAttention_70695161692392
// MI455X (gfx1250) — compile-verified, matrix operand order checked by reading, not run on hardware
//
#include <hip/hip_runtime.h>
#include <hip/hip_bf16.h>

typedef __attribute__((ext_vector_type(16))) __bf16 v16bf;
typedef __attribute__((ext_vector_type(8)))  float  v8f;

#define B_  2
#define H_  16
#define S_  2048
#define D_  1024
#define DK_ 64

union Frag { v16bf v; uint4 q[2]; };

// ---- CDNA5 async global->LDS copy (16 bytes per lane), tracked by ASYNCcnt.
__device__ __forceinline__ void async_b128(void* lds, const void* g) {
    unsigned lofs = (unsigned)(unsigned long long)(size_t)lds;   // low 32 bits = LDS byte offset
    unsigned long long ga = (unsigned long long)(size_t)g;
    asm volatile("global_load_async_to_lds_b128 %0, %1, off"
                 :: "v"(lofs), "v"(ga) : "memory");
}

// ---------------------------------------------------------------------------
// GEMM: C[M,N] = A[M,K] @ W[N,K]^T + bias ; M=4096, N=K=1024 ; BK=64
// ABF: 0 = A is fp32, 1 = A is bf16 (async-staged)
// MODE: 0 = out bf16 [B,H,S,64] ; 1 = out bf16 [B,H,64,S] ; 2 = out fp32 [M,N]
// ---------------------------------------------------------------------------
template<int ABF, int MODE>
__global__ __launch_bounds__(256)
void gemm_qkv(const void* __restrict__ Ap, const float* __restrict__ W,
              const float* __restrict__ bias, void* __restrict__ outp)
{
    constexpr int K = 1024, N = 1024;
    __shared__ __bf16 As[128][72];   // 64 data + 8 pad elems: conflict-free frag reads
    __shared__ __bf16 Bs[128][72];

    const int t    = threadIdx.x;
    const int lane = t & 31, wave = t >> 5;
    const int wm   = wave >> 1, wn = wave & 1;      // 4 x 2 wave grid
    const int tile_n = blockIdx.x & 7;              // N/128 = 8
    const int tile_m = blockIdx.x >> 3;             // M/128 = 32
    const int m0 = tile_m * 128, n0 = tile_n * 128;

    const int lrow = t >> 1;                        // 0..127
    const int lcb  = (t & 1) * 32;                  // 0 or 32
    const int hl   = lane >> 4, ll = lane & 15;

    v8f acc[2][4];
    #pragma unroll
    for (int i = 0; i < 2; i++)
        #pragma unroll
        for (int j = 0; j < 4; j++)
            #pragma unroll
            for (int e = 0; e < 8; e++) acc[i][j][e] = 0.0f;

    for (int k0 = 0; k0 < K; k0 += 64) {
        __syncthreads();
        // stage A tile (row = lrow, 32 elements starting at lcb)
        if constexpr (ABF) {
            const __bf16* g = (const __bf16*)Ap + (size_t)(m0 + lrow) * K + k0 + lcb;
            __bf16* l = &As[lrow][lcb];
            #pragma unroll
            for (int j = 0; j < 4; j++) async_b128(l + j * 8, g + j * 8);
        } else {
            union { __bf16 h[32]; uint4 q[4]; } u;
            const float4* p = reinterpret_cast<const float4*>(
                (const float*)Ap + (size_t)(m0 + lrow) * K + k0 + lcb);
            #pragma unroll
            for (int v = 0; v < 8; v++) {
                float4 f = p[v];
                u.h[v*4+0] = (__bf16)f.x; u.h[v*4+1] = (__bf16)f.y;
                u.h[v*4+2] = (__bf16)f.z; u.h[v*4+3] = (__bf16)f.w;
            }
            #pragma unroll
            for (int j = 0; j < 4; j++)
                *reinterpret_cast<uint4*>(&As[lrow][lcb + j*8]) = u.q[j];
        }
        // stage W tile (always fp32 -> bf16)
        {
            union { __bf16 h[32]; uint4 q[4]; } u;
            const float4* p = reinterpret_cast<const float4*>(
                W + (size_t)(n0 + lrow) * K + k0 + lcb);
            #pragma unroll
            for (int v = 0; v < 8; v++) {
                float4 f = p[v];
                u.h[v*4+0] = (__bf16)f.x; u.h[v*4+1] = (__bf16)f.y;
                u.h[v*4+2] = (__bf16)f.z; u.h[v*4+3] = (__bf16)f.w;
            }
            #pragma unroll
            for (int j = 0; j < 4; j++)
                *reinterpret_cast<uint4*>(&Bs[lrow][lcb + j*8]) = u.q[j];
        }
        if constexpr (ABF) asm volatile("s_wait_asynccnt 0" ::: "memory");
        __syncthreads();

        #pragma unroll
        for (int kk = 0; kk < 2; kk++) {
            const int kc = kk * 32 + hl * 16;   // lanes 16-31 hold K=16..31
            Frag a[2], b[4];
            #pragma unroll
            for (int mt = 0; mt < 2; mt++) {
                const uint4* p = reinterpret_cast<const uint4*>(&As[wm*32 + mt*16 + ll][kc]);
                a[mt].q[0] = p[0]; a[mt].q[1] = p[1];
            }
            #pragma unroll
            for (int nt = 0; nt < 4; nt++) {
                const uint4* p = reinterpret_cast<const uint4*>(&Bs[wn*64 + nt*16 + ll][kc]);
                b[nt].q[0] = p[0]; b[nt].q[1] = p[1];
            }
            #pragma unroll
            for (int mt = 0; mt < 2; mt++)
                #pragma unroll
                for (int nt = 0; nt < 4; nt++)
                    acc[mt][nt] = __builtin_amdgcn_wmma_f32_16x16x32_bf16(
                        false, a[mt].v, false, b[nt].v, (short)0, acc[mt][nt], false, false);
        }
    }

    // epilogue: bias + layout-specific store
    #pragma unroll
    for (int nt = 0; nt < 4; nt++) {
        const int gcol = n0 + wn*64 + nt*16 + ll;
        const float bv = bias[gcol];
        #pragma unroll
        for (int mt = 0; mt < 2; mt++) {
            #pragma unroll
            for (int r = 0; r < 8; r++) {
                const int grow = m0 + wm*32 + mt*16 + r + hl*8;  // C layout rows
                const float val = acc[mt][nt][r] + bv;
                if constexpr (MODE == 2) {
                    ((float*)outp)[(size_t)grow * N + gcol] = val;
                } else {
                    const int bb = grow >> 11;       // / S_
                    const int ss = grow & (S_ - 1);
                    const int hh = gcol >> 6;        // / 64
                    const int dk = gcol & 63;
                    size_t idx;
                    if constexpr (MODE == 0)
                        idx = (((size_t)bb * H_ + hh) * S_ + ss) * DK_ + dk;
                    else
                        idx = (((size_t)bb * H_ + hh) * DK_ + dk) * S_ + ss;
                    ((__bf16*)outp)[idx] = (__bf16)val;
                }
            }
        }
    }
}

// ---------------------------------------------------------------------------
// Flash attention: 4 waves x 16 query rows per block; 64-key blocks streamed
// through double-buffered LDS with async global->LDS loads (ASYNCcnt).
// Q,K bf16 [b,h,s,64]; V bf16 transposed [b,h,64,s]; out ctx bf16 [b,s,1024]
// ---------------------------------------------------------------------------
__global__ __launch_bounds__(128)
void attn_fwd(const void* __restrict__ Qp, const void* __restrict__ Kp,
              const void* __restrict__ Vp, void* __restrict__ Cp)
{
    const __bf16* Q  = (const __bf16*)Qp;
    const __bf16* Kb = (const __bf16*)Kp;
    const __bf16* Vt = (const __bf16*)Vp;
    __bf16* Ctx      = (__bf16*)Cp;

    __shared__ __bf16 Ks[2][64][72];   // double-buffered keys (s x d), padded
    __shared__ __bf16 Vs[2][64][72];   // double-buffered V^T (d x s), padded
    __shared__ __bf16 Ps[4][16][72];   // per-wave P relayout (C -> A operand)

    const int t    = threadIdx.x;
    const int lane = t & 31, wave = t >> 5;
    const int hl   = lane >> 4, ll = lane & 15;

    const int qb = blockIdx.x & 31;          // S/64 = 32 query tiles
    const int h  = (blockIdx.x >> 5) & 15;
    const int b  = blockIdx.x >> 9;

    const size_t headQ = ((size_t)b * H_ + h) * S_ * DK_;
    const size_t headV = ((size_t)b * H_ + h) * DK_ * S_;

    const int crow = t >> 1;         // 0..63 cooperative-load row
    const int ccb  = (t & 1) * 32;   // 0 or 32 (elements)

    // Q fragments for this wave's 16 rows (kept in registers all along)
    Frag qf[2];
    #pragma unroll
    for (int kf = 0; kf < 2; kf++) {
        const int row = qb*64 + wave*16 + ll;
        const int d   = kf*32 + hl*16;
        const uint4* p = reinterpret_cast<const uint4*>(Q + headQ + (size_t)row * DK_ + d);
        qf[kf].q[0] = p[0]; qf[kf].q[1] = p[1];
    }

    float mrow[8], lrow[8];
    #pragma unroll
    for (int r = 0; r < 8; r++) { mrow[r] = -3.0e38f; lrow[r] = 0.0f; }
    v8f accO[4];
    #pragma unroll
    for (int nt = 0; nt < 4; nt++)
        #pragma unroll
        for (int e = 0; e < 8; e++) accO[nt][e] = 0.0f;

    // issue a K/V block into LDS buffer `buf` (8 async b128 per thread)
    auto issue_block = [&](int kb, int buf) {
        const __bf16* gk = Kb + headQ + (size_t)(kb + crow) * DK_ + ccb;
        __bf16* lk = &Ks[buf][crow][ccb];
        #pragma unroll
        for (int j = 0; j < 4; j++) async_b128(lk + j*8, gk + j*8);
        const __bf16* gv = Vt + headV + (size_t)crow * S_ + kb + ccb;
        __bf16* lv = &Vs[buf][crow][ccb];
        #pragma unroll
        for (int j = 0; j < 4; j++) async_b128(lv + j*8, gv + j*8);
    };

    constexpr int NB = S_ / 64;   // 32 key blocks
    issue_block(0, 0);            // prologue

    for (int i = 0; i < NB; i++) {
        const int cur = i & 1;
        if (i + 1 < NB) {
            issue_block((i + 1) * 64, 1 - cur);                 // prefetch next block
            asm volatile("s_wait_asynccnt 8" ::: "memory");     // wait for *older* 8
        } else {
            asm volatile("s_wait_asynccnt 0" ::: "memory");
        }
        __syncthreads();   // current buffer complete for all waves

        // scores S = Q @ K^T  (16 x 64 tile per wave)
        v8f accS[4];
        #pragma unroll
        for (int nt = 0; nt < 4; nt++)
            #pragma unroll
            for (int e = 0; e < 8; e++) accS[nt][e] = 0.0f;
        #pragma unroll
        for (int kf = 0; kf < 2; kf++) {
            #pragma unroll
            for (int nt = 0; nt < 4; nt++) {
                Frag bfr;
                const uint4* p = reinterpret_cast<const uint4*>(
                    &Ks[cur][nt*16 + ll][kf*32 + hl*16]);
                bfr.q[0] = p[0]; bfr.q[1] = p[1];
                accS[nt] = __builtin_amdgcn_wmma_f32_16x16x32_bf16(
                    false, qf[kf].v, false, bfr.v, (short)0, accS[nt], false, false);
            }
        }

        // online softmax (C layout: VGPR r -> row r (lanes 0-15) / r+8 (lanes 16-31))
        #pragma unroll
        for (int r = 0; r < 8; r++) {
            float sv[4]; float smax = -3.0e38f;
            #pragma unroll
            for (int nt = 0; nt < 4; nt++) {
                sv[nt] = accS[nt][r] * 0.125f;           // 1/sqrt(64)
                smax = fmaxf(smax, sv[nt]);
            }
            #pragma unroll
            for (int off = 1; off < 16; off <<= 1)       // reduce within 16-lane half
                smax = fmaxf(smax, __shfl_xor(smax, off, 32));
            const float mnew = fmaxf(mrow[r], smax);
            const float corr = __expf(mrow[r] - mnew);
            mrow[r] = mnew;
            lrow[r] *= corr;
            #pragma unroll
            for (int nt = 0; nt < 4; nt++) accO[nt][r] *= corr;
            #pragma unroll
            for (int nt = 0; nt < 4; nt++) {
                const float pv = __expf(sv[nt] - mnew);
                lrow[r] += pv;                            // per-lane partial sum
                Ps[wave][r + hl*8][nt*16 + ll] = (__bf16)pv;
            }
        }
        asm volatile("s_wait_dscnt 0" ::: "memory");      // P writes visible in-wave

        // O += P @ V
        #pragma unroll
        for (int kf = 0; kf < 2; kf++) {
            Frag pf;
            const uint4* pp = reinterpret_cast<const uint4*>(&Ps[wave][ll][kf*32 + hl*16]);
            pf.q[0] = pp[0]; pf.q[1] = pp[1];
            #pragma unroll
            for (int nt = 0; nt < 4; nt++) {
                Frag vf;
                const uint4* pv = reinterpret_cast<const uint4*>(
                    &Vs[cur][nt*16 + ll][kf*32 + hl*16]);
                vf.q[0] = pv[0]; vf.q[1] = pv[1];
                accO[nt] = __builtin_amdgcn_wmma_f32_16x16x32_bf16(
                    false, pf.v, false, vf.v, (short)0, accO[nt], false, false);
            }
        }
        __syncthreads();   // all waves done reading `cur` before it is refilled
    }

    // finalize: reduce row sums, normalize, write ctx [b, s, h*64 + d]
    #pragma unroll
    for (int r = 0; r < 8; r++) {
        float lsum = lrow[r];
        #pragma unroll
        for (int off = 1; off < 16; off <<= 1) lsum += __shfl_xor(lsum, off, 32);
        const float rinv = 1.0f / lsum;
        const int grow = qb*64 + wave*16 + r + hl*8;
        #pragma unroll
        for (int nt = 0; nt < 4; nt++) {
            const int gcol = h*64 + nt*16 + ll;
            Ctx[((size_t)b * S_ + grow) * D_ + gcol] = (__bf16)(accO[nt][r] * rinv);
        }
    }
}

// ---------------------------------------------------------------------------
extern "C" void kernel_launch(void* const* d_in, const int* in_sizes, int n_in,
                              void* d_out, int out_size, void* d_ws, size_t ws_size,
                              hipStream_t stream) {
    const float* q_in = (const float*)d_in[0];
    const float* k_in = (const float*)d_in[1];
    const float* v_in = (const float*)d_in[2];
    const float* w_q  = (const float*)d_in[3];
    const float* b_q  = (const float*)d_in[4];
    const float* w_k  = (const float*)d_in[5];
    const float* b_k  = (const float*)d_in[6];
    const float* w_v  = (const float*)d_in[7];
    const float* b_v  = (const float*)d_in[8];
    const float* w_o  = (const float*)d_in[9];
    const float* b_o  = (const float*)d_in[10];

    char* ws = (char*)d_ws;
    const size_t headBytes = (size_t)B_ * H_ * S_ * DK_ * 2;  // 8 MB each (bf16)
    void* Qws = (void*)(ws);
    void* Kws = (void*)(ws + headBytes);
    void* Vws = (void*)(ws + 2 * headBytes);
    void* Cws = (void*)(ws + 3 * headBytes);

    dim3 gemmGrid(256), gemmBlock(256);
    gemm_qkv<0, 0><<<gemmGrid, gemmBlock, 0, stream>>>(q_in, w_q, b_q, Qws);
    gemm_qkv<0, 0><<<gemmGrid, gemmBlock, 0, stream>>>(k_in, w_k, b_k, Kws);
    gemm_qkv<0, 1><<<gemmGrid, gemmBlock, 0, stream>>>(v_in, w_v, b_v, Vws);

    attn_fwd<<<dim3(B_ * H_ * (S_ / 64)), dim3(128), 0, stream>>>(Qws, Kws, Vws, Cws);

    gemm_qkv<1, 2><<<gemmGrid, gemmBlock, 0, stream>>>(Cws, w_o, b_o, d_out);
}
